// GAT_edge_4_41137196761629
// MI455X (gfx1250) — matrix operand order repaired and checked by reading, not compiled, not run on hardware
//
#include <hip/hip_runtime.h>
#include <math.h>

#define NN   20000
#define EE   640000
#define DD   128
#define HIDD 128
#define BB   64
#define EPSF 1e-5f

typedef __attribute__((ext_vector_type(16))) _Float16 v16h;
typedef __attribute__((ext_vector_type(8)))  float    v8f;
typedef __attribute__((ext_vector_type(4)))  float    f4raw;

union HF16 { v16h v; f4raw q[2]; };   // 16 halves <-> 2x (4 floats raw bits)

// ---------- helpers ----------
__device__ __forceinline__ unsigned fenc(float x) {
    unsigned u = __float_as_uint(x);
    return (u & 0x80000000u) ? ~u : (u | 0x80000000u);
}
__device__ __forceinline__ float fdec(unsigned k) {
    unsigned u = (k & 0x80000000u) ? (k & 0x7FFFFFFFu) : ~k;
    return __uint_as_float(u);
}
__device__ __forceinline__ float sigf(float x) { return 1.0f / (1.0f + expf(-x)); }

// ---------- B fragment pre-pack: per-lane order, contiguous k = kc*32 + 16*(lane/16) + j as for A, f16 ----------
// fragment index t = ntile*128 + kchunk*32 + lane ; 16 contiguous halves each
__global__ void pack_bfrag(const float* __restrict__ B, int ldb,
                           _Float16* __restrict__ Bfrag, int Ntiles) {
    int t = blockIdx.x * blockDim.x + threadIdx.x;
    if (t >= Ntiles * 128) return;
    const int lane  = t & 31;
    const int kc    = (t >> 5) & 3;
    const int ntile = t >> 7;
    const int k0    = kc * 32 + (lane >> 4) * 16;
    const int col   = ntile * 16 + (lane & 15);
    _Float16* o = Bfrag + (size_t)t * 16;
#pragma unroll
    for (int j = 0; j < 16; ++j)
        o[j] = (_Float16)B[(size_t)(k0 + j) * ldb + col];
}

// pack [W | skip_W] (each 128x128) directly into fragments for Ncols=256
__global__ void pack_bfrag_layer(const float* __restrict__ W, const float* __restrict__ Sk,
                                 _Float16* __restrict__ Bfrag) {
    int t = blockIdx.x * blockDim.x + threadIdx.x;
    if (t >= 16 * 128) return;
    const int lane  = t & 31;
    const int kc    = (t >> 5) & 3;
    const int ntile = t >> 7;
    const int k0    = kc * 32 + (lane >> 4) * 16;
    const int col   = ntile * 16 + (lane & 15);
    const float* S  = (col < 128) ? (W + col) : (Sk + (col - 128));
    _Float16* o = Bfrag + (size_t)t * 16;
#pragma unroll
    for (int j = 0; j < 16; ++j)
        o[j] = (_Float16)S[(size_t)(k0 + j) * 128];
}

// ---------- WMMA GEMM: C[M,16*Ntiles] = A[M,128] * B, f32 A, prepacked f16 B ----------
// grid.x = M/16 tiles; grid.y * (blockDim/32) covers Ntiles
__global__ void wmma_gemm128(const float* __restrict__ A, int lda,
                             const _Float16* __restrict__ Bfrag,
                             float* __restrict__ C, int ldc, int Ntiles) {
    const int lane = threadIdx.x & 31;
    const int wave = threadIdx.x >> 5;
    const int wpb  = blockDim.x >> 5;
    const int ntile = blockIdx.y * wpb + wave;
    if (ntile >= Ntiles) return;                 // wave-uniform exit, EXEC stays full
    const int mrow  = blockIdx.x * 16 + (lane & 15);
    const int koffA = (lane >> 4) * 16;          // A: lanes 0-15 K=+0..15, 16-31 K=+16..31
    const int col   = ntile * 16 + (lane & 15);
    const float* Arow = A + (size_t)mrow * lda + koffA;
    const _Float16* bp = Bfrag + ((size_t)ntile * 128 + lane) * 16;
    v8f acc = {};
#pragma unroll
    for (int k0 = 0; k0 < 128; k0 += 32) {
        const f4raw a0 = *reinterpret_cast<const f4raw*>(Arow + k0);
        const f4raw a1 = *reinterpret_cast<const f4raw*>(Arow + k0 + 4);
        const f4raw a2 = *reinterpret_cast<const f4raw*>(Arow + k0 + 8);
        const f4raw a3 = *reinterpret_cast<const f4raw*>(Arow + k0 + 12);
        HF16 b;
        b.q[0] = *reinterpret_cast<const f4raw*>(bp);
        b.q[1] = *reinterpret_cast<const f4raw*>(bp + 8);
        bp += 32 * 16;                           // next k-chunk's fragments
        v16h af;
#pragma unroll
        for (int j = 0; j < 4; ++j) {
            af[j]      = (_Float16)a0[j];
            af[4 + j]  = (_Float16)a1[j];
            af[8 + j]  = (_Float16)a2[j];
            af[12 + j] = (_Float16)a3[j];
        }
        acc = __builtin_amdgcn_wmma_f32_16x16x32_f16(false, af, false, b.v,
                                                     (short)0, acc, false, false);
    }
#pragma unroll
    for (int j = 0; j < 8; ++j) {
        const int r = blockIdx.x * 16 + j + (lane >> 4) * 8;
        C[(size_t)r * ldc + col] = acc[j];
    }
}

// ---------- build V_all[128,32]: per-layer W_e folded with att_edge ----------
__global__ void build_vall(const float* __restrict__ We0, const float* __restrict__ ae0,
                           const float* __restrict__ We1, const float* __restrict__ ae1,
                           const float* __restrict__ We2, const float* __restrict__ ae2,
                           const float* __restrict__ We3, const float* __restrict__ ae3,
                           float* __restrict__ V) {
    int t = blockIdx.x * blockDim.x + threadIdx.x;
    if (t >= 128 * 32) return;
    int d = t >> 5, col = t & 31;
    float acc = 0.0f;
    if (col < 24) {
        int l = col >> 3, h = col & 7;
        const float* We = (l == 0) ? We0 : (l == 1) ? We1 : We2;
        const float* ae = (l == 0) ? ae0 : (l == 1) ? ae1 : ae2;
#pragma unroll
        for (int c = 0; c < 16; ++c)
            acc += We[d * 128 + h * 16 + c] * ae[h * 16 + c];
    } else if (col == 24) {
        for (int c = 0; c < 128; ++c)
            acc += We3[d * 128 + c] * ae3[c];
    }
    V[d * 32 + col] = acc;
}

// ---------- per-node attention coefficients a_src/a_dst ----------
__global__ void node_att(const float* __restrict__ HS, const float* __restrict__ att_s,
                         const float* __restrict__ att_d, float* __restrict__ a_src,
                         float* __restrict__ a_dst, int HSH, int total) {
    int t = blockIdx.x * blockDim.x + threadIdx.x;
    if (t >= total) return;
    const int H = 1 << HSH;
    const int C = 128 >> HSH;
    const int n = t >> HSH, h = t & (H - 1);
    const float* row = HS + (size_t)n * 256 + h * C;
    const float* as = att_s + h * C;
    const float* ad = att_d + h * C;
    float s = 0.0f, d2 = 0.0f;
    for (int c = 0; c < C; ++c) { float v = row[c]; s += v * as[c]; d2 += v * ad[c]; }
    a_src[t] = s; a_dst[t] = d2;
}

// ---------- edge alpha + segment-max (over dst) ----------
__global__ void edge_alpha_max(const int* __restrict__ src, const int* __restrict__ dst,
                               const float* __restrict__ a_src, const float* __restrict__ a_dst,
                               const float* __restrict__ a_e_all, int colbase, int HSH,
                               float* __restrict__ alphab, unsigned* __restrict__ mkeys,
                               int total) {
    int t = blockIdx.x * blockDim.x + threadIdx.x;
    if (t >= total) return;
    const int H = 1 << HSH;
    const int e = t >> HSH, h = t & (H - 1);
    const int s = src[e], d = dst[e];
    float a = a_src[s * H + h] + a_dst[d * H + h] + a_e_all[(size_t)e * 32 + colbase + h];
    a = (a > 0.0f) ? a : 0.2f * a;                   // leaky relu, slope 0.2
    alphab[t] = a;
    atomicMax(mkeys + d * H + h, fenc(a));
}

// ---------- exp + segment-sum ----------
__global__ void edge_exp_sum(const int* __restrict__ dst, float* __restrict__ alphab,
                             const unsigned* __restrict__ mkeys, float* __restrict__ den,
                             int HSH, int total) {
    int t = blockIdx.x * blockDim.x + threadIdx.x;
    if (t >= total) return;
    const int H = 1 << HSH;
    const int e = t >> HSH, h = t & (H - 1);
    const int d = dst[e];
    float ex = expf(alphab[t] - fdec(mkeys[d * H + h]));
    alphab[t] = ex;
    atomicAdd(den + d * H + h, ex);
}

// ---------- weighted message scatter: one wave per edge ----------
__global__ void edge_scatter(const int* __restrict__ src, const int* __restrict__ dst,
                             const float* __restrict__ HS, const float* __restrict__ ex,
                             const float* __restrict__ den, float* __restrict__ outacc,
                             int H, int E) {
    int gw = (blockIdx.x * blockDim.x + threadIdx.x) >> 5;
    int lane = threadIdx.x & 31;
    if (gw >= E) return;
    const int s = src[gw], d = dst[gw];
    const int c = lane * 4;
    const int h = (c >> 4) & (H - 1);
    const float w = ex[(size_t)gw * H + h] / den[(size_t)d * H + h];
    const float4 hv = *reinterpret_cast<const float4*>(HS + (size_t)s * 256 + c);
    float* po = outacc + (size_t)d * 128 + c;
    atomicAdd(po + 0, hv.x * w);
    atomicAdd(po + 1, hv.y * w);
    atomicAdd(po + 2, hv.z * w);
    atomicAdd(po + 3, hv.w * w);
}

// ---------- node epilogue: +bias -> BN -> LN -> +skip -> ELU ; one wave per node ----------
__global__ void node_epilogue(const float* __restrict__ outacc, const float* __restrict__ HS,
                              const float* __restrict__ bias, const float* __restrict__ bng,
                              const float* __restrict__ bnb, const float* __restrict__ bnm,
                              const float* __restrict__ bnv, const float* __restrict__ lng,
                              const float* __restrict__ lnb, const float* __restrict__ skb,
                              float* __restrict__ xnext, int N) {
    int gw = (blockIdx.x * blockDim.x + threadIdx.x) >> 5;
    int lane = threadIdx.x & 31;
    if (gw >= N) return;
    const int c0 = lane * 4;
    const float4 v = *reinterpret_cast<const float4*>(outacc + (size_t)gw * 128 + c0);
    float vals[4] = {v.x, v.y, v.z, v.w};
    float s = 0.0f, sq = 0.0f;
#pragma unroll
    for (int i = 0; i < 4; ++i) {
        const int c = c0 + i;
        float t = vals[i] + bias[c];
        t = (t - bnm[c]) * rsqrtf(bnv[c] + EPSF) * bng[c] + bnb[c];
        vals[i] = t; s += t; sq += t * t;
    }
#pragma unroll
    for (int off = 16; off > 0; off >>= 1) {
        s  += __shfl_xor(s,  off, 32);
        sq += __shfl_xor(sq, off, 32);
    }
    const float mu  = s * (1.0f / 128.0f);
    const float var = sq * (1.0f / 128.0f) - mu * mu;
    const float inv = rsqrtf(var + EPSF);
#pragma unroll
    for (int i = 0; i < 4; ++i) {
        const int c = c0 + i;
        float y = (vals[i] - mu) * inv * lng[c] + lnb[c];
        y += HS[(size_t)gw * 256 + 128 + c] + skb[c];        // skip branch
        y = (y > 0.0f) ? y : (expf(y) - 1.0f);               // ELU
        xnext[(size_t)gw * 128 + c] = y;
    }
}

// ---------- Set2Set ----------
__global__ void s2s_gates(const float* __restrict__ qstar, const float* __restrict__ hs,
                          const float* __restrict__ Wih, const float* __restrict__ Whh,
                          const float* __restrict__ bih, const float* __restrict__ bhh,
                          float* __restrict__ gates) {
    int t = blockIdx.x * blockDim.x + threadIdx.x;
    if (t >= BB * 512) return;
    int b = t >> 9, j = t & 511;
    float acc = bih[j] + bhh[j];
    const float* q = qstar + b * 256;
    const float* h = hs + b * 128;
    for (int k = 0; k < 256; ++k) acc += q[k] * Wih[k * 512 + j];
    for (int k = 0; k < 128; ++k) acc += h[k] * Whh[k * 512 + j];
    gates[t] = acc;
}

__global__ void s2s_cell(const float* __restrict__ gates, float* __restrict__ cs,
                         float* __restrict__ hs) {
    int t = blockIdx.x * blockDim.x + threadIdx.x;
    if (t >= BB * 128) return;
    int b = t >> 7, j = t & 127;
    const float* g = gates + b * 512;
    float gi = g[j], gf = g[128 + j], gg = g[256 + j], go = g[384 + j];
    float c = sigf(gf) * cs[t] + sigf(gi) * tanhf(gg);
    cs[t] = c;
    hs[t] = sigf(go) * tanhf(c);
}

__global__ void s2s_dot_max(const float* __restrict__ x, const float* __restrict__ hs,
                            const int* __restrict__ batch, float* __restrict__ ebuf,
                            unsigned* __restrict__ mk, int N) {
    int gw = (blockIdx.x * blockDim.x + threadIdx.x) >> 5;
    int lane = threadIdx.x & 31;
    if (gw >= N) return;
    const int b = batch[gw];
    const float4 xv = *reinterpret_cast<const float4*>(x + (size_t)gw * 128 + lane * 4);
    const float4 qv = *reinterpret_cast<const float4*>(hs + (size_t)b * 128 + lane * 4);
    float s = xv.x * qv.x + xv.y * qv.y + xv.z * qv.z + xv.w * qv.w;
#pragma unroll
    for (int off = 16; off > 0; off >>= 1) s += __shfl_xor(s, off, 32);
    if (lane == 0) { ebuf[gw] = s; atomicMax(mk + b, fenc(s)); }
}

__global__ void s2s_expsum(float* __restrict__ ebuf, const int* __restrict__ batch,
                           const unsigned* __restrict__ mk, float* __restrict__ den, int N) {
    int t = blockIdx.x * blockDim.x + threadIdx.x;
    if (t >= N) return;
    const int b = batch[t];
    float ex = expf(ebuf[t] - fdec(mk[b]));
    ebuf[t] = ex;
    atomicAdd(den + b, ex);
}

__global__ void s2s_r(const float* __restrict__ x, const float* __restrict__ ebuf,
                      const int* __restrict__ batch, const float* __restrict__ den,
                      float* __restrict__ r, int N) {
    int gw = (blockIdx.x * blockDim.x + threadIdx.x) >> 5;
    int lane = threadIdx.x & 31;
    if (gw >= N) return;
    const int b = batch[gw];
    const float a = ebuf[gw] / den[b];
    const int c = lane * 4;
    const float4 xv = *reinterpret_cast<const float4*>(x + (size_t)gw * 128 + c);
    float* pr = r + (size_t)b * 128 + c;
    atomicAdd(pr + 0, a * xv.x);
    atomicAdd(pr + 1, a * xv.y);
    atomicAdd(pr + 2, a * xv.z);
    atomicAdd(pr + 3, a * xv.w);
}

__global__ void s2s_qstar(const float* __restrict__ hs, const float* __restrict__ r,
                          float* __restrict__ qstar) {
    int t = blockIdx.x * blockDim.x + threadIdx.x;
    if (t >= BB * 256) return;
    int b = t >> 8, j = t & 255;
    qstar[t] = (j < 128) ? hs[b * 128 + j] : r[b * 128 + (j - 128)];
}

__global__ void s2s_proj(const float* __restrict__ qstar, const float* __restrict__ pW,
                         const float* __restrict__ pb, float* __restrict__ out) {
    int t = blockIdx.x * blockDim.x + threadIdx.x;
    if (t >= BB * 128) return;
    int b = t >> 7, j = t & 127;
    float acc = pb[j];
    const float* q = qstar + b * 256;
    for (int k = 0; k < 256; ++k) acc += q[k] * pW[k * 128 + j];
    out[t] = fmaxf(acc, 0.0f);
}

// ---------- host-side orchestration ----------
extern "C" void kernel_launch(void* const* d_in, const int* in_sizes, int n_in,
                              void* d_out, int out_size, void* d_ws, size_t ws_size,
                              hipStream_t stream) {
    const float* x_in  = (const float*)d_in[0];
    const float* ea    = (const float*)d_in[1];
    const int*   ei    = (const int*)d_in[2];
    const int*   src   = ei;
    const int*   dstv  = ei + EE;
    const int*   batch = (const int*)d_in[3];
    auto LPf = [&](int l, int k) -> const float* { return (const float*)d_in[4 + l * 14 + k]; };
    // k: 0 W, 1 W_e, 2 att_src, 3 att_dst, 4 att_edge, 5 bias, 6 bn_g, 7 bn_b,
    //    8 bn_m, 9 bn_v, 10 ln_g, 11 ln_b, 12 skip_W, 13 skip_b
    const float* Wih = (const float*)d_in[60];
    const float* Whh = (const float*)d_in[61];
    const float* bih = (const float*)d_in[62];
    const float* bhh = (const float*)d_in[63];
    const float* pW  = (const float*)d_in[64];
    const float* pb  = (const float*)d_in[65];

    // workspace carve-out (float units; every block multiple of 16 floats -> 64B aligned)
    float* w = (float*)d_ws;
    float*    a_e_all = w; w += (size_t)EE * 32;
    float*    HS      = w; w += (size_t)NN * 256;
    float*    xb0     = w; w += (size_t)NN * 128;
    float*    xb1     = w; w += (size_t)NN * 128;
    float*    outacc  = w; w += (size_t)NN * 128;
    float*    a_srcb  = w; w += (size_t)NN * 8;
    float*    a_dstb  = w; w += (size_t)NN * 8;
    unsigned* mkeys   = (unsigned*)w; w += (size_t)NN * 8;
    float*    denb    = w; w += (size_t)NN * 8;
    float*    alphab  = w; w += (size_t)EE * 8;
    float*    Vall    = w; w += 128 * 32;
    _Float16* Bfrag   = (_Float16*)w; w += 16 * 128 * 16 / 2;   // 32768 halves
    float*    gates   = w; w += BB * 512;
    float*    hsb     = w; w += BB * 128;
    float*    csb     = w; w += BB * 128;
    float*    qstar   = w; w += BB * 256;
    float*    rbuf    = w; w += BB * 128;
    float*    ebuf    = w; w += NN;
    unsigned* mk2     = (unsigned*)w; w += BB;
    float*    den2    = w; w += BB;

    // folded edge-attention projection: a_e_all = ea @ V_all  (one pass over edge_attr)
    build_vall<<<(128 * 32 + 255) / 256, 256, 0, stream>>>(
        LPf(0, 1), LPf(0, 4), LPf(1, 1), LPf(1, 4),
        LPf(2, 1), LPf(2, 4), LPf(3, 1), LPf(3, 4), Vall);
    pack_bfrag<<<1, 256, 0, stream>>>(Vall, 32, Bfrag, 2);
    wmma_gemm128<<<dim3(EE / 16, 1), 64, 0, stream>>>(ea, 128, Bfrag, a_e_all, 32, 2);

    const float* xcur = x_in;
    float* xbufs[2] = {xb0, xb1};
    for (int l = 0; l < 4; ++l) {
        const int H = (l < 3) ? 8 : 1;
        const int HSH = (l < 3) ? 3 : 0;
        const int colbase = (l < 3) ? l * 8 : 24;

        pack_bfrag_layer<<<(16 * 128 + 255) / 256, 256, 0, stream>>>(LPf(l, 0), LPf(l, 12), Bfrag);
        // HS[:,0:128] = x@W (messages), HS[:,128:256] = x@skip_W
        wmma_gemm128<<<dim3(NN / 16, 2), 256, 0, stream>>>(xcur, 128, Bfrag, HS, 256, 16);

        node_att<<<(NN * H + 255) / 256, 256, 0, stream>>>(
            HS, LPf(l, 2), LPf(l, 3), a_srcb, a_dstb, HSH, NN * H);

        hipMemsetAsync(mkeys, 0, (size_t)NN * 8 * 4, stream);   // 0 == encoded minimum
        hipMemsetAsync(denb, 0, (size_t)NN * 8 * 4, stream);
        hipMemsetAsync(outacc, 0, (size_t)NN * 128 * 4, stream);

        edge_alpha_max<<<(EE * H + 255) / 256, 256, 0, stream>>>(
            src, dstv, a_srcb, a_dstb, a_e_all, colbase, HSH, alphab, mkeys, EE * H);
        edge_exp_sum<<<(EE * H + 255) / 256, 256, 0, stream>>>(
            dstv, alphab, mkeys, denb, HSH, EE * H);
        edge_scatter<<<EE / 8, 256, 0, stream>>>(src, dstv, HS, alphab, denb, outacc, H, EE);

        float* xn = xbufs[l & 1];
        node_epilogue<<<NN / 8, 256, 0, stream>>>(
            outacc, HS, LPf(l, 5), LPf(l, 6), LPf(l, 7), LPf(l, 8), LPf(l, 9),
            LPf(l, 10), LPf(l, 11), LPf(l, 13), xn, NN);
        xcur = xn;
    }

    // Set2Set (3 steps)
    hipMemsetAsync(qstar, 0, BB * 256 * 4, stream);
    hipMemsetAsync(hsb, 0, BB * 128 * 4, stream);
    hipMemsetAsync(csb, 0, BB * 128 * 4, stream);
    for (int s = 0; s < 3; ++s) {
        s2s_gates<<<(BB * 512 + 255) / 256, 256, 0, stream>>>(qstar, hsb, Wih, Whh, bih, bhh, gates);
        s2s_cell<<<(BB * 128 + 255) / 256, 256, 0, stream>>>(gates, csb, hsb);
        hipMemsetAsync(mk2, 0, BB * 4, stream);
        hipMemsetAsync(den2, 0, BB * 4, stream);
        hipMemsetAsync(rbuf, 0, BB * 128 * 4, stream);
        s2s_dot_max<<<NN / 8, 256, 0, stream>>>(xcur, hsb, batch, ebuf, mk2, NN);
        s2s_expsum<<<(NN + 255) / 256, 256, 0, stream>>>(ebuf, batch, mk2, den2, NN);
        s2s_r<<<NN / 8, 256, 0, stream>>>(xcur, ebuf, batch, den2, rbuf, NN);
        s2s_qstar<<<(BB * 256 + 255) / 256, 256, 0, stream>>>(hsb, rbuf, qstar);
    }
    s2s_proj<<<(BB * 128 + 255) / 256, 256, 0, stream>>>(qstar, pW, pb, (float*)d_out);
}
